// Rs_GCN_70712341561912
// MI455X (gfx1250) — matrix operand order repaired and checked by reading, not compiled, not run on hardware
//
#include <hip/hip_runtime.h>
#include <math.h>

#define B_  4
#define C_  512
#define CI_ 256
#define N_  4096
#define EPS_ 1e-5f
#define SEGS_ 64

typedef __attribute__((ext_vector_type(16))) __bf16 v16bf;
typedef __attribute__((ext_vector_type(8)))  __bf16 v8bf;
typedef __attribute__((ext_vector_type(8)))  float  v8f;

// ---------------------------------------------------------------------------
// WMMA fragment loaders (bf16, 16x16x32). Layouts per CDNA5 ISA §7.12.2:
//  A (16x32, MxK): lane<16 -> row=lane,   K chunks [0..7] and [16..23]
//                  lane>=16 -> row=lane-16, K chunks [8..15] and [24..31]
//  B (32x16, KxN): lane holds column n=lane&15; same K chunks as A:
//                  lanes 0-15 K=[0..7],[16..23], lanes 16-31 K=[8..15],[24..31]
// ---------------------------------------------------------------------------
__device__ __forceinline__ v16bf load_a_frag(const __bf16* __restrict__ A,
                                             int lda, int kk, int lane) {
    int r  = lane & 15;
    int hi = lane >> 4;
    const __bf16* p = A + (size_t)r * lda + kk + hi * 8;
    v8bf lo = *(const v8bf*)(p);
    v8bf hh = *(const v8bf*)(p + 16);
    return __builtin_shufflevector(lo, hh, 0,1,2,3,4,5,6,7,8,9,10,11,12,13,14,15);
}

__device__ __forceinline__ v16bf load_b_frag(const __bf16* __restrict__ Bt,
                                             int ldb, int kk, int lane) {
    int r  = lane & 15;
    int hi = lane >> 4;
    const __bf16* p = Bt + (size_t)r * ldb + kk + hi * 8;
    v8bf lo = *(const v8bf*)(p);
    v8bf hh = *(const v8bf*)(p + 16);
    return __builtin_shufflevector(lo, hh, 0,1,2,3,4,5,6,7,8,9,10,11,12,13,14,15);
}

// ---------------------------------------------------------------------------
// Batched GEMM with transposed (column-major) output:
//   Dt[z][n][m] = sum_k A[z][m][k] * Bt[z][n][k] * scale (+ bias)
// A: [M,K] row-major, Bt: [Nn,K] row-major, Dt: [Nn,M] row-major.
// The D-matrix VGPR layout (vgpr r, lane L -> M = r + 8*(L>=16), N = L&15)
// makes each lane's 8 accumulator values CONTIGUOUS in Dt -> one packed
// 16B (bf16) or 32B (f32) store per 16x16 tile per lane.
// Wave tile 32x64 (2x4 accumulators), 8 waves/block -> 64x256 tile.
// BIAS_MODE: 0 = none, 1 = bias[m], 2 = bias[n]
// ---------------------------------------------------------------------------
template<int BIAS_MODE, bool OUT_BF16>
__global__ void __launch_bounds__(256)
gemm_bf16_wmma_t(const __bf16* __restrict__ A, const __bf16* __restrict__ Bt,
                 const float* __restrict__ bias, void* __restrict__ D,
                 int M, int Nn, int K, float scale,
                 size_t sA, size_t sB, size_t sD)
{
    constexpr int BM = 2, BN = 4;
    const int lane = threadIdx.x & 31;
    const int wid  = threadIdx.x >> 5;
    const int m_base = blockIdx.y * 64  + (wid >> 2) * 32;
    const int n_base = blockIdx.x * 256 + (wid & 3) * 64;

    const __bf16* Ab  = A  + sA * blockIdx.z;
    const __bf16* Btb = Bt + sB * blockIdx.z;

    v8f acc[BM][BN] = {};

    for (int kk = 0; kk < K; kk += 32) {
        v16bf a[BM];
        v16bf b[BN];
#pragma unroll
        for (int i = 0; i < BM; ++i)
            a[i] = load_a_frag(Ab + (size_t)(m_base + 16 * i) * K, K, kk, lane);
#pragma unroll
        for (int j = 0; j < BN; ++j)
            b[j] = load_b_frag(Btb + (size_t)(n_base + 16 * j) * K, K, kk, lane);
#pragma unroll
        for (int i = 0; i < BM; ++i)
#pragma unroll
            for (int j = 0; j < BN; ++j)
                acc[i][j] = __builtin_amdgcn_wmma_f32_16x16x32_bf16(
                    false, a[i], false, b[j], (short)0, acc[i][j], false, false);
    }

    const int hi = lane >> 4;
    const int nc = lane & 15;

    float bnv[BN] = {};
    if constexpr (BIAS_MODE == 2) {
#pragma unroll
        for (int j = 0; j < BN; ++j) bnv[j] = bias[n_base + 16 * j + nc];
    }

#pragma unroll
    for (int i = 0; i < BM; ++i) {
        const int mb = m_base + 16 * i + hi * 8;      // 8 consecutive M values
        v8f bmv = {};
        if constexpr (BIAS_MODE == 1) bmv = *(const v8f*)(bias + mb);
#pragma unroll
        for (int j = 0; j < BN; ++j) {
            const int n = n_base + 16 * j + nc;
            if constexpr (OUT_BF16) {
                v8bf ov;
#pragma unroll
                for (int r = 0; r < 8; ++r) {
                    float val = acc[i][j][r] * scale;
                    if constexpr (BIAS_MODE == 1) val += bmv[r];
                    if constexpr (BIAS_MODE == 2) val += bnv[j];
                    ov[r] = (__bf16)val;
                }
                *(v8bf*)((__bf16*)D + sD * blockIdx.z + (size_t)n * M + mb) = ov;
            } else {
                v8f ov;
#pragma unroll
                for (int r = 0; r < 8; ++r) {
                    float val = acc[i][j][r] * scale;
                    if constexpr (BIAS_MODE == 1) val += bmv[r];
                    if constexpr (BIAS_MODE == 2) val += bnv[j];
                    ov[r] = val;
                }
                *(v8f*)((float*)D + sD * blockIdx.z + (size_t)n * M + mb) = ov;
            }
        }
    }
}

// ---------------------------------------------------------------------------
// v [B,C,N] f32  ->  vT [B,N,C] bf16  (LDS-tiled 32x32 transpose + convert)
// ---------------------------------------------------------------------------
__global__ void __launch_bounds__(256)
transpose_cvt_bf16(const float* __restrict__ v, __bf16* __restrict__ vT)
{
    __shared__ float tile[32][33];
    const int b  = blockIdx.z;
    const int c0 = blockIdx.y * 32;
    const int n0 = blockIdx.x * 32;
    const float* src = v  + (size_t)b * C_ * N_;
    __bf16*      dst = vT + (size_t)b * N_ * C_;
#pragma unroll
    for (int p = 0; p < 4; ++p) {
        int idx = threadIdx.x + p * 256;
        int row = idx >> 5, col = idx & 31;           // row: c, col: n
        tile[row][col] = src[(size_t)(c0 + row) * N_ + n0 + col];
    }
    __syncthreads();
#pragma unroll
    for (int p = 0; p < 4; ++p) {
        int idx = threadIdx.x + p * 256;
        int row = idx >> 5, col = idx & 31;           // row: n, col: c
        dst[(size_t)(n0 + row) * C_ + c0 + col] = (__bf16)tile[col][row];
    }
}

__global__ void cvt_f32_bf16(const float* __restrict__ in,
                             __bf16* __restrict__ out, int n)
{
    int i = blockIdx.x * blockDim.x + threadIdx.x;
    if (i < n) out[i] = (__bf16)in[i];
}

// ---------------------------------------------------------------------------
// BN stats stage 1: WyT [B,N,C] (channel-contiguous, fully coalesced reads).
// grid (C/64, SEGS); each block: 64 channels x 256 rows; thread owns one
// channel for 64 rows, then 4-way LDS reduce.
// ---------------------------------------------------------------------------
__global__ void __launch_bounds__(256)
bn_partial_kernel(const float* __restrict__ WyT, float* __restrict__ partial)
{
    const int c   = blockIdx.x * 64 + (threadIdx.x & 63);
    const int rg  = threadIdx.x >> 6;                 // 0..3
    const int seg = blockIdx.y;
    constexpr int ROWS = (B_ * N_) / SEGS_;           // 256
    const int row0 = seg * ROWS;
    float s = 0.f, s2 = 0.f;
    for (int p = 0; p < ROWS / 4; ++p) {
        float x = WyT[(size_t)(row0 + rg + p * 4) * C_ + c];
        s += x; s2 += x * x;
    }
    __shared__ float sm[256];
    __shared__ float sq[256];
    sm[threadIdx.x] = s; sq[threadIdx.x] = s2;
    __syncthreads();
    if (threadIdx.x < 64) {
        float ts = sm[threadIdx.x] + sm[threadIdx.x + 64] +
                   sm[threadIdx.x + 128] + sm[threadIdx.x + 192];
        float tq = sq[threadIdx.x] + sq[threadIdx.x + 64] +
                   sq[threadIdx.x + 128] + sq[threadIdx.x + 192];
        partial[(size_t)seg * (2 * C_) + c]      = ts;
        partial[(size_t)seg * (2 * C_) + C_ + c] = tq;
    }
}

__global__ void __launch_bounds__(256)
bn_finalize_kernel(const float* __restrict__ partial, float* __restrict__ stats)
{
    const int c = blockIdx.x * 256 + threadIdx.x;
    float s = 0.f, s2 = 0.f;
    for (int seg = 0; seg < SEGS_; ++seg) {
        s  += partial[(size_t)seg * (2 * C_) + c];
        s2 += partial[(size_t)seg * (2 * C_) + C_ + c];
    }
    const float inv = 1.0f / (float)(B_ * N_);
    stats[c]      = s  * inv;                          // mean
    stats[C_ + c] = s2 * inv;                          // E[x^2]
}

// ---------------------------------------------------------------------------
// BN apply + affine + residual, fused with LDS-tiled transpose back to [B,C,N]:
//   out[b,c,n] = (WyT[b,n,c]-mean)*rsqrt(var+eps)*gamma + beta + v[b,c,n]
// ---------------------------------------------------------------------------
__global__ void __launch_bounds__(256)
bn_apply_t_kernel(const float* __restrict__ WyT, const float* __restrict__ v,
                  const float* __restrict__ stats, const float* __restrict__ gamma,
                  const float* __restrict__ beta, float* __restrict__ out)
{
    __shared__ float tile[32][33];
    const int b  = blockIdx.z;
    const int c0 = blockIdx.y * 32;
    const int n0 = blockIdx.x * 32;
    const float* src = WyT + (size_t)b * N_ * C_;
#pragma unroll
    for (int p = 0; p < 4; ++p) {
        int idx = threadIdx.x + p * 256;
        int row = idx >> 5, col = idx & 31;           // row: n, col: c (coalesced)
        tile[row][col] = src[(size_t)(n0 + row) * C_ + c0 + col];
    }
    __syncthreads();
#pragma unroll
    for (int p = 0; p < 4; ++p) {
        int idx = threadIdx.x + p * 256;
        int row = idx >> 5, col = idx & 31;           // row: c, col: n (coalesced)
        const int c = c0 + row;
        float mean   = stats[c];
        float var    = stats[C_ + c] - mean * mean;
        float invstd = rsqrtf(var + EPS_);
        size_t o = (size_t)b * C_ * N_ + (size_t)c * N_ + n0 + col;
        out[o] = (tile[col][row] - mean) * invstd * gamma[c] + beta[c] + v[o];
    }
}

// ---------------------------------------------------------------------------
extern "C" void kernel_launch(void* const* d_in, const int* in_sizes, int n_in,
                              void* d_out, int out_size, void* d_ws, size_t ws_size,
                              hipStream_t stream)
{
    (void)in_sizes; (void)n_in; (void)out_size; (void)ws_size;
    const float* v     = (const float*)d_in[0];
    const float* Wg    = (const float*)d_in[1];
    const float* bg    = (const float*)d_in[2];
    const float* Wth   = (const float*)d_in[3];
    const float* bth   = (const float*)d_in[4];
    const float* Wph   = (const float*)d_in[5];
    const float* bph   = (const float*)d_in[6];
    const float* Ww    = (const float*)d_in[7];
    const float* bw    = (const float*)d_in[8];
    const float* gamma = (const float*)d_in[9];
    const float* beta  = (const float*)d_in[10];

    char* ws = (char*)d_ws;
    size_t off = 0;
    auto alloc = [&](size_t bytes) -> char* {
        off = (off + 255) & ~(size_t)255;
        char* p = ws + off;
        off += bytes;
        return p;
    };

    __bf16* vT    = (__bf16*)alloc((size_t)B_ * N_ * C_  * 2);  // [B,N,C]
    __bf16* WgB   = (__bf16*)alloc((size_t)CI_ * C_ * 2);
    __bf16* WthB  = (__bf16*)alloc((size_t)CI_ * C_ * 2);
    __bf16* WphB  = (__bf16*)alloc((size_t)CI_ * C_ * 2);
    __bf16* WwB   = (__bf16*)alloc((size_t)C_ * CI_ * 2);
    __bf16* gP    = (__bf16*)alloc((size_t)B_ * CI_ * N_ * 2); // [B,Ci,N]
    __bf16* phiP  = (__bf16*)alloc((size_t)B_ * CI_ * N_ * 2); // [B,Ci,N]
    __bf16* thT   = (__bf16*)alloc((size_t)B_ * N_ * CI_ * 2); // [B,N,Ci]
    __bf16* S     = (__bf16*)alloc((size_t)B_ * CI_ * CI_ * 2);// [B,Ci(k),Ci(c)]
    __bf16* T     = (__bf16*)alloc((size_t)B_ * C_ * CI_ * 2); // [B,C(o),Ci(k)]
    float*  WyT   = (float*) alloc((size_t)B_ * N_ * C_ * 4);  // [B,N,C]
    float*  part  = (float*) alloc((size_t)SEGS_ * 2 * C_ * 4);
    float*  stats = (float*) alloc((size_t)2 * C_ * 4);

    // 0) precision conversion + transpose of v
    transpose_cvt_bf16<<<dim3(N_ / 32, C_ / 32, B_), 256, 0, stream>>>(v, vT);
    const int wsz = CI_ * C_;
    cvt_f32_bf16<<<(wsz + 255) / 256, 256, 0, stream>>>(Wg,  WgB,  wsz);
    cvt_f32_bf16<<<(wsz + 255) / 256, 256, 0, stream>>>(Wth, WthB, wsz);
    cvt_f32_bf16<<<(wsz + 255) / 256, 256, 0, stream>>>(Wph, WphB, wsz);
    cvt_f32_bf16<<<(C_ * CI_ + 255) / 256, 256, 0, stream>>>(Ww, WwB, C_ * CI_);

    const dim3 blk(256);
    // 1) g/phi projections oriented A=vT, Bt=W  -> Dt = [Ci,N] packed stores.
    //    bias indexed by GEMM column (ci) -> BIAS_MODE 2.
    gemm_bf16_wmma_t<2, true><<<dim3(CI_ / 256, N_ / 64, B_), blk, 0, stream>>>(
        vT, WgB, bg, gP, N_, CI_, C_, 1.0f,
        (size_t)N_ * C_, 0, (size_t)CI_ * N_);
    gemm_bf16_wmma_t<2, true><<<dim3(CI_ / 256, N_ / 64, B_), blk, 0, stream>>>(
        vT, WphB, bph, phiP, N_, CI_, C_, 1.0f,
        (size_t)N_ * C_, 0, (size_t)CI_ * N_);
    //    theta oriented A=W, Bt=vT -> Dt = [N,Ci] packed; bias per GEMM row.
    gemm_bf16_wmma_t<1, true><<<dim3(N_ / 256, CI_ / 64, B_), blk, 0, stream>>>(
        WthB, vT, bth, thT, CI_, N_, C_, 1.0f,
        0, (size_t)N_ * C_, (size_t)N_ * CI_);

    // 2) S[k][c] = sum_m phi[k,m] g[c,m] / N  (A=gP m=c, Bt=phiP n=k; Dt packed)
    gemm_bf16_wmma_t<0, true><<<dim3(CI_ / 256, CI_ / 64, B_), blk, 0, stream>>>(
        gP, phiP, nullptr, S, CI_, CI_, N_, 1.0f / (float)N_,
        (size_t)CI_ * N_, (size_t)CI_ * N_, (size_t)CI_ * CI_);

    // 3) T[o][k] = sum_c S[k,c] Ww[o,c]  (A=S m=k, Bt=Ww n=o; Dt packed)
    gemm_bf16_wmma_t<0, true><<<dim3(C_ / 256, CI_ / 64, B_), blk, 0, stream>>>(
        S, WwB, nullptr, T, CI_, C_, CI_, 1.0f,
        (size_t)CI_ * CI_, 0, (size_t)C_ * CI_);

    // 4) WyT[n][o] = sum_k T[o,k] th[n,k] + bw[o]  (A=T m=o, Bt=thT n=n)
    gemm_bf16_wmma_t<1, false><<<dim3(N_ / 256, C_ / 64, B_), blk, 0, stream>>>(
        T, thT, bw, WyT, C_, N_, CI_, 1.0f,
        (size_t)C_ * CI_, (size_t)N_ * CI_, (size_t)N_ * C_);

    // 5) BatchNorm (training-mode biased stats) + affine + residual
    bn_partial_kernel<<<dim3(C_ / 64, SEGS_), 256, 0, stream>>>(WyT, part);
    bn_finalize_kernel<<<dim3(C_ / 256), 256, 0, stream>>>(part, stats);
    bn_apply_t_kernel<<<dim3(N_ / 32, C_ / 32, B_), 256, 0, stream>>>(
        WyT, v, stats, gamma, beta, (float*)d_out);
}
